// TreeBlock_38053410242790
// MI455X (gfx1250) — matrix operand order repaired and checked by reading, not compiled, not run on hardware
//
#include <hip/hip_runtime.h>

typedef __attribute__((ext_vector_type(16))) __bf16   v16bf;
typedef __attribute__((ext_vector_type(2)))  __bf16   v2bf;
typedef __attribute__((ext_vector_type(8)))  float    v8f;
typedef __attribute__((ext_vector_type(4)))  unsigned v4u;
typedef __attribute__((ext_vector_type(4)))  float    v4f;

#define TILE_R 4
#define TILE_C 32
#define Y1R 6
#define Y1C 34
#define NPIX_REAL 204   /* 6*34 */
#define NT1 13          /* ceil(204/16) */

#define C_IN  256
#define C_MID 128
#define C_OUT 256
#define IMG_H 128
#define IMG_W 128
#define HW    (IMG_H * IMG_W)

/* ---- LDS layout (bytes) ---- */
#define WREG_OFF 0
#define WREG_SZ  69632            /* max(w1 128*528, wt 256*272, w2 fp32 18432) */
#define W1_STR   528              /* 256 bf16 + 16B pad  -> conflict-free A-frag */
#define WT_STR   272              /* 128 bf16 + 16B pad */
#define Y1_OFF   WREG_SZ
#define Y1_STR   432              /* [ch][px]: 208 bf16 px + pad */
#define Y1_SZ    (128 * Y1_STR)   /* 55296 */
#define Y2_OFF   (Y1_OFF + Y1_SZ) /* 124928 */
#define Y2_STR   272              /* [px][ch]: 128 bf16 ch + pad */
#define Y2_SZ    (128 * Y2_STR)   /* 34816 */
#define SMEM_SZ  (Y2_OFF + Y2_SZ) /* 159744 */

__global__ __launch_bounds__(256) void fused_treeblock(
    const float* __restrict__ x,  const float* __restrict__ w1,
    const float* __restrict__ b1, const float* __restrict__ w2,
    const float* __restrict__ b2, const float* __restrict__ wt,
    const float* __restrict__ bt, float* __restrict__ out)
{
  __shared__ __align__(16) char smem[SMEM_SZ];
  const int tid  = threadIdx.x;
  const int lane = tid & 31;
  const int wv   = tid >> 5;
  const int l15  = lane & 15;
  const bool hi  = lane >= 16;

  const int wg = blockIdx.x;
  const int n  = wg >> 7;          /* 128 blocks per image */
  const int hb = (wg >> 2) & 31;
  const int cb = wg & 3;
  const int h0 = hb * TILE_R;
  const int c0 = cb * TILE_C;

  /* ================= stage w1 fp32 -> bf16 LDS ================= */
  for (int i = tid; i < (C_MID * C_IN) / 2; i += 256) {
    int ch = i >> 7, kp = (i & 127) << 1;
    v2bf pk;
    pk[0] = (__bf16)w1[ch * C_IN + kp];
    pk[1] = (__bf16)w1[ch * C_IN + kp + 1];
    *(v2bf*)(smem + WREG_OFF + ch * W1_STR + kp * 2) = pk;
  }
  __syncthreads();

  /* ============ phase 1: 1x1 conv 256->128 (bf16 WMMA) ============ */
  const float* ximg = x + (size_t)n * C_IN * HW;
  for (int t = wv; t < NT1; t += 8) {
    int p   = t * 16 + l15;
    int r   = p / Y1C;
    int cc  = p - r * Y1C;
    int row = h0 - 1 + r;
    int col = c0 - 1 + cc;
    bool valid = (p < NPIX_REAL) && (row >= 0) && (row < IMG_H) &&
                 (col >= 0) && (col < IMG_W);
    const float* xp = ximg + (valid ? (row * IMG_W + col) : 0);

    v8f acc[8];
    v8f zero = {0,0,0,0,0,0,0,0};
#pragma unroll
    for (int m = 0; m < 8; ++m) acc[m] = zero;

    for (int kk = 0; kk < 8; ++kk) {
      int chb = kk * 32 + (hi ? 16 : 0);
      v16bf bfr;
#pragma unroll
      for (int j = 0; j < 16; ++j)
        bfr[j] = (__bf16)xp[(size_t)(chb + j) * HW];
#pragma unroll
      for (int m = 0; m < 8; ++m) {
        const char* ap = smem + WREG_OFF + (m * 16 + l15) * W1_STR +
                         kk * 64 + (hi ? 32 : 0);
        union { v4u q[2]; v16bf v; } afr;
        afr.q[0] = *(const v4u*)ap;
        afr.q[1] = *(const v4u*)(ap + 16);
        acc[m] = __builtin_amdgcn_wmma_f32_16x16x32_bf16(
            false, afr.v, false, bfr, (short)0, acc[m], false, false);
      }
    }
    /* bias + bf16 store into y1 [ch][px]; zero pad pixels (SAME padding) */
#pragma unroll
    for (int m = 0; m < 8; ++m) {
      int ch0 = m * 16 + (hi ? 8 : 0);
      v4f bA = *(const v4f*)(b1 + ch0);
      v4f bB = *(const v4f*)(b1 + ch0 + 4);
      float bias[8] = {bA.x, bA.y, bA.z, bA.w, bB.x, bB.y, bB.z, bB.w};
#pragma unroll
      for (int v = 0; v < 8; ++v) {
        float val = valid ? (acc[m][v] + bias[v]) : 0.0f;
        *(__bf16*)(smem + Y1_OFF + (ch0 + v) * Y1_STR + p * 2) = (__bf16)val;
      }
    }
  }
  __syncthreads();

  /* ================= stage w2 fp32 over weight region ================= */
  float* w2l = (float*)(smem + WREG_OFF);
  for (int i = tid; i < C_MID * 36; i += 256) w2l[i] = w2[i];
  __syncthreads();

  /* ============ phase 2: grouped 3x3 conv (fp32 VALU) ============ */
  for (int it = 0; it < 32; ++it) {
    int i   = tid + (it << 8);
    int ch  = i >> 6;
    int pp  = i & 63;
    int rr  = pp >> 4;
    int cpl = (pp & 15) << 1;      /* pixel pair base column */
    const float* wp = w2l + ch * 36;
    float bias = b2[ch];
    float a0 = bias, a1 = bias;
    int gch = ch & ~3;
#pragma unroll
    for (int ig = 0; ig < 4; ++ig) {
      int c1 = gch + ig;
#pragma unroll
      for (int dy = 0; dy < 3; ++dy) {
        const char* yp = smem + Y1_OFF + c1 * Y1_STR +
                         ((rr + dy) * Y1C + cpl) * 2;
        v2bf p0 = *(const v2bf*)yp;
        v2bf p1 = *(const v2bf*)(yp + 4);
        float y0  = (float)p0[0];
        float y1v = (float)p0[1];
        float y2v = (float)p1[0];
        float y3  = (float)p1[1];
        float q0 = wp[ig * 9 + dy * 3 + 0];
        float q1 = wp[ig * 9 + dy * 3 + 1];
        float q2 = wp[ig * 9 + dy * 3 + 2];
        a0 = fmaf(q0, y0,  fmaf(q1, y1v, fmaf(q2, y2v, a0)));
        a1 = fmaf(q0, y1v, fmaf(q1, y2v, fmaf(q2, y3,  a1)));
      }
    }
    int px0 = rr * TILE_C + cpl;
    *(__bf16*)(smem + Y2_OFF + px0 * Y2_STR + ch * 2)       = (__bf16)a0;
    *(__bf16*)(smem + Y2_OFF + (px0 + 1) * Y2_STR + ch * 2) = (__bf16)a1;
  }
  __syncthreads();

  /* ================= stage wt fp32 -> bf16 LDS ================= */
  for (int i = tid; i < (C_OUT * C_MID) / 2; i += 256) {
    int och = i >> 6, kp = (i & 63) << 1;
    v2bf pk;
    pk[0] = (__bf16)wt[och * C_MID + kp];
    pk[1] = (__bf16)wt[och * C_MID + kp + 1];
    *(v2bf*)(smem + WREG_OFF + och * WT_STR + kp * 2) = pk;
  }
  __syncthreads();

  /* ====== phase 3: tail 1x1 128->256 (WMMA) + bias + residual ====== */
  {
    int p3  = wv * 16 + l15;        /* one N-tile per wave */
    int rr3 = p3 >> 5, cc3 = p3 & 31;
    v8f acc[16];
    v8f zero = {0,0,0,0,0,0,0,0};
#pragma unroll
    for (int m = 0; m < 16; ++m) acc[m] = zero;

    for (int kk = 0; kk < 4; ++kk) {
      const char* bp = smem + Y2_OFF + p3 * Y2_STR + kk * 64 + (hi ? 32 : 0);
      union { v4u q[2]; v16bf v; } bfr;
      bfr.q[0] = *(const v4u*)bp;
      bfr.q[1] = *(const v4u*)(bp + 16);
#pragma unroll
      for (int m = 0; m < 16; ++m) {
        const char* ap = smem + WREG_OFF + (m * 16 + l15) * WT_STR +
                         kk * 64 + (hi ? 32 : 0);
        union { v4u q[2]; v16bf v; } afr;
        afr.q[0] = *(const v4u*)ap;
        afr.q[1] = *(const v4u*)(ap + 16);
        acc[m] = __builtin_amdgcn_wmma_f32_16x16x32_bf16(
            false, afr.v, false, bfr.v, (short)0, acc[m], false, false);
      }
    }
    size_t obase = (size_t)n * C_OUT * HW +
                   (size_t)(h0 + rr3) * IMG_W + (c0 + cc3);
#pragma unroll
    for (int m = 0; m < 16; ++m) {
      int ch0 = m * 16 + (hi ? 8 : 0);
      v4f tA = *(const v4f*)(bt + ch0);
      v4f tB = *(const v4f*)(bt + ch0 + 4);
      float bias[8] = {tA.x, tA.y, tA.z, tA.w, tB.x, tB.y, tB.z, tB.w};
#pragma unroll
      for (int v = 0; v < 8; ++v) {
        size_t gi = obase + (size_t)(ch0 + v) * HW;
        out[gi] = acc[m][v] + bias[v] + x[gi];
      }
    }
  }
}

extern "C" void kernel_launch(void* const* d_in, const int* in_sizes, int n_in,
                              void* d_out, int out_size, void* d_ws, size_t ws_size,
                              hipStream_t stream) {
  (void)in_sizes; (void)n_in; (void)out_size; (void)d_ws; (void)ws_size;
  const float* x  = (const float*)d_in[0];
  const float* w1 = (const float*)d_in[1];
  const float* b1 = (const float*)d_in[2];
  const float* w2 = (const float*)d_in[3];
  const float* b2 = (const float*)d_in[4];
  const float* wt = (const float*)d_in[5];
  const float* bt = (const float*)d_in[6];
  float* out = (float*)d_out;
  /* 8 images * 32 row-blocks * 4 col-blocks = 1024 workgroups */
  fused_treeblock<<<dim3(1024), dim3(256), 0, stream>>>(x, w1, b1, w2, b2, wt, bt, out);
}
